// KimiLinearBlock_72885595013413
// MI455X (gfx1250) — matrix operand order repaired and checked by reading, not compiled, not run on hardware
//
#include <hip/hip_runtime.h>
#include <cstddef>
#include <cstdint>

typedef unsigned int u32;
typedef __bf16 bf16;
typedef bf16  v16bf __attribute__((ext_vector_type(16)));
typedef float v8f   __attribute__((ext_vector_type(8)));

#define LN_EPS 1e-5f
#define USE_ASYNC_LDS 1

// fast sigmoid/silu: v_exp_f32 + v_rcp_f32 (no IEEE div chain)
__device__ __forceinline__ float sigm_f(float x) {
    return __builtin_amdgcn_rcpf(1.0f + __expf(-x));
}
__device__ __forceinline__ float silu_f(float x) { return x * sigm_f(x); }

// ---------------------------------------------------------------------------
// fp32 -> bf16 conversion (grid-stride)
// ---------------------------------------------------------------------------
__global__ void cvt_f32_bf16(const float* __restrict__ in, bf16* __restrict__ out, size_t n) {
    size_t i = (size_t)blockIdx.x * blockDim.x + threadIdx.x;
    size_t stride = (size_t)gridDim.x * blockDim.x;
    for (; i < n; i += stride) out[i] = (bf16)in[i];
}

// ---------------------------------------------------------------------------
// LayerNorm over D=1024, output bf16. One block (256 thr) per row.
// ---------------------------------------------------------------------------
__global__ __launch_bounds__(256)
void layernorm_bf16(const float* __restrict__ x, const float* __restrict__ w,
                    const float* __restrict__ b, bf16* __restrict__ out) {
    const int row = blockIdx.x;
    const int t = threadIdx.x;
    const float* xr = x + (size_t)row * 1024;
    float v0[4];
    float s = 0.f, s2 = 0.f;
#pragma unroll
    for (int i = 0; i < 4; i++) {
        float v = xr[t + 256 * i];
        v0[i] = v; s += v; s2 += v * v;
    }
    __shared__ float rs[256], rs2[256];
    rs[t] = s; rs2[t] = s2;
    __syncthreads();
    for (int off = 128; off > 0; off >>= 1) {
        if (t < off) { rs[t] += rs[t + off]; rs2[t] += rs2[t + off]; }
        __syncthreads();
    }
    const float mu  = rs[0] * (1.0f / 1024.0f);
    const float var = rs2[0] * (1.0f / 1024.0f) - mu * mu;
    const float inv = rsqrtf(var + LN_EPS);
    bf16* orow = out + (size_t)row * 1024;
#pragma unroll
    for (int i = 0; i < 4; i++) {
        int c = t + 256 * i;
        orow[c] = (bf16)((v0[i] - mu) * inv * w[c] + b[c]);
    }
}

// ---------------------------------------------------------------------------
// WMMA bf16 GEMM:  C[M,N] = A[M,K](bf16) * W[K,N](bf16), fp32 accumulate.
// Block tile 64(M) x 128(N), BK=32. 8 waves: 2(M) x 4(N), wave tile 32x32
// = 2x2 v_wmma_f32_16x16x32_bf16 accumulators.
// A tile streamed via global_load_async_to_lds_b128 (ASYNCcnt); B tile
// zipped into K-pair dwords through VGPRs. Padded LDS strides de-conflict
// the fragment gathers.
// EPI: 0=f32  1=silu f32  2=sigmoid f32  3=bf16  4=GLU(aux bf16)*silu->bf16
//      5=aux(f32)+acc -> f32
// ---------------------------------------------------------------------------
#define LA_WORDS (64 * 17)
#define LB_WORDS (16 * 132)

template <int EPI>
__global__ __launch_bounds__(256)
void gemm_bf16_wmma(const bf16* __restrict__ A, const bf16* __restrict__ Bw,
                    float* __restrict__ Cf, bf16* __restrict__ Cb,
                    const void* __restrict__ aux, int M, int N, int K) {
    // single static LDS block: lA at group-segment offset 0 (needed by the
    // async-copy asm which takes a raw LDS byte offset), lB right after.
    __shared__ u32 lds[LA_WORDS + LB_WORDS];
    u32* lA = lds;                 // [m][Kpair] stride 17 (+1 pad)
    u32* lB = lds + LA_WORDS;      // [Kpair][n] stride 132

    const int t     = threadIdx.x;
    const int lane  = t & 31;
    const int wave  = t >> 5;
    const int waveM = wave & 1;    // 0..1 -> 32-row slice
    const int waveN = wave >> 1;   // 0..3 -> 32-col slice
    const int blockN = blockIdx.x * 128;
    const int blockM = blockIdx.y * 64;

    const u32* Au = (const u32*)A; // bf16 pair view, pairs along K
    const int  Kp = K >> 1;

    // cooperative-load coordinates
    const int am  = t >> 2;          // 0..63 row of A tile
    const int au0 = (t & 3) << 2;    // uint offset within 16-uint row
    const int bkp = t >> 4;          // 0..15 K-pair row of B tile
    const int bn0 = (t & 15) << 3;   // 0..120 col group

    const u32 laddrA = (u32)((am * 17 + au0) * 4);  // LDS byte offset of this
                                                    // thread's 16B A chunk

    v8f acc[2][2];
#pragma unroll
    for (int i = 0; i < 2; i++)
#pragma unroll
        for (int j = 0; j < 2; j++)
            acc[i][j] = (v8f){0.f, 0.f, 0.f, 0.f, 0.f, 0.f, 0.f, 0.f};

    union Frag { u32 u[8]; v16bf v; };

    for (int k0 = 0; k0 < K; k0 += 32) {
        // ---- B global loads for this K-slab (overlap with prev compute) ----
        const u32* agp = Au + (size_t)(blockM + am) * Kp + (k0 >> 1) + au0;
        const bf16* b0p = Bw + (size_t)(k0 + 2 * bkp) * N + blockN + bn0;
        const uint4 be = *(const uint4*)b0p;       // K even row, 8 bf16
        const uint4 bo = *(const uint4*)(b0p + N); // K odd  row, 8 bf16
#if !USE_ASYNC_LDS
        const uint4 av = *(const uint4*)agp;
#endif

        if (k0 + 32 < K) {  // prefetch next slab into near caches
            __builtin_prefetch(agp + 16, 0, 3);
            __builtin_prefetch(b0p + (size_t)32 * N, 0, 3);
        }

        __syncthreads();  // previous iteration's LDS reads complete
#if USE_ASYNC_LDS
        {   // A tile: direct global -> LDS async copy, 16B per lane
            const unsigned long long ga = (unsigned long long)(uintptr_t)agp;
            asm volatile("global_load_async_to_lds_b128 %0, %1, off"
                         :: "v"(laddrA), "v"(ga) : "memory");
        }
#else
        lA[am * 17 + au0 + 0] = av.x;
        lA[am * 17 + au0 + 1] = av.y;
        lA[am * 17 + au0 + 2] = av.z;
        lA[am * 17 + au0 + 3] = av.w;
#endif
        // B tile: zip even/odd K rows into K-pair dwords (lo=K even, hi=K odd)
        u32* lbp = lB + bkp * 132 + bn0;
        lbp[0] = (be.x & 0xFFFFu) | (bo.x << 16);
        lbp[1] = (be.x >> 16)     | (bo.x & 0xFFFF0000u);
        lbp[2] = (be.y & 0xFFFFu) | (bo.y << 16);
        lbp[3] = (be.y >> 16)     | (bo.y & 0xFFFF0000u);
        lbp[4] = (be.z & 0xFFFFu) | (bo.z << 16);
        lbp[5] = (be.z >> 16)     | (bo.z & 0xFFFF0000u);
        lbp[6] = (be.w & 0xFFFFu) | (bo.w << 16);
        lbp[7] = (be.w >> 16)     | (bo.w & 0xFFFF0000u);
#if USE_ASYNC_LDS
        asm volatile("s_wait_asynccnt 0x0" ::: "memory");
#endif
        __syncthreads();

        // ---- build fragments per documented VGPR layouts ----
        const int g  = lane >> 4;   // half-wave
        const int ml = lane & 15;
        Frag afr[2], bfr[2];
#pragma unroll
        for (int mi = 0; mi < 2; mi++) {
            const u32* ar = lA + (waveM * 32 + mi * 16 + ml) * 17;
#pragma unroll
            for (int vv = 0; vv < 8; vv++) {
                const int p = ((vv >= 4) ? 8 : 0) + g * 4 + (vv & 3);  // K-pair idx
                afr[mi].u[vv] = ar[p];
            }
        }
#pragma unroll
        for (int ni = 0; ni < 2; ni++) {
            const int col = waveN * 32 + ni * 16 + ml;
#pragma unroll
            for (int vv = 0; vv < 8; vv++)
                bfr[ni].u[vv] = lB[(((vv >= 4) ? 8 : 0) + g * 4 + (vv & 3)) * 132 + col];
        }

#pragma unroll
        for (int mi = 0; mi < 2; mi++)
#pragma unroll
            for (int ni = 0; ni < 2; ni++)
                acc[mi][ni] = __builtin_amdgcn_wmma_f32_16x16x32_bf16(
                    /*neg_a=*/false, afr[mi].v, /*neg_b=*/false, bfr[ni].v,
                    /*c_mod=*/(short)0, acc[mi][ni],
                    /*reuse_a=*/false, /*reuse_b=*/false);
    }

    // ---- epilogue: D layout = lane n = L%16, rows m = (L/16)*8 + r ----
#pragma unroll
    for (int mi = 0; mi < 2; mi++)
#pragma unroll
        for (int ni = 0; ni < 2; ni++)
#pragma unroll
            for (int r = 0; r < 8; r++) {
                const int row = blockM + waveM * 32 + mi * 16 + (lane >> 4) * 8 + r;
                const int col = blockN + waveN * 32 + ni * 16 + (lane & 15);
                const size_t idx = (size_t)row * N + col;
                const float v = acc[mi][ni][r];
                if constexpr (EPI == 0) Cf[idx] = v;
                else if constexpr (EPI == 1) Cf[idx] = silu_f(v);
                else if constexpr (EPI == 2) Cf[idx] = sigm_f(v);
                else if constexpr (EPI == 3) Cb[idx] = (bf16)v;
                else if constexpr (EPI == 4)
                    Cb[idx] = (bf16)((float)((const bf16*)aux)[idx] * silu_f(v));
                else if constexpr (EPI == 5)
                    Cf[idx] = ((const float*)aux)[idx] + v;
            }
}

// ---------------------------------------------------------------------------
// beta = sigmoid(h @ Wb): tiny N=16 GEMM, one thread per (token, head).
// ---------------------------------------------------------------------------
__global__ void beta_kernel(const bf16* __restrict__ hb, const bf16* __restrict__ WbB,
                            float* __restrict__ beta, int total) {
    int id = blockIdx.x * blockDim.x + threadIdx.x;
    if (id >= total) return;
    const int tok = id >> 4, h = id & 15;
    const bf16* hr = hb + (size_t)tok * 1024;
    float s = 0.f;
    for (int k = 0; k < 1024; k++)
        s = fmaf((float)hr[k], (float)WbB[k * 16 + h], s);
    beta[id] = sigm_f(s);
}

// ---------------------------------------------------------------------------
// L2-normalize k per (token, head) over dk=64, in place.
// ---------------------------------------------------------------------------
__global__ void l2norm_kernel(float* __restrict__ k) {
    const int j = threadIdx.x;                       // 0..63
    const size_t base = (size_t)blockIdx.x * 64;     // blockIdx = token*16 + head
    float v = k[base + j];
    __shared__ float red[64];
    red[j] = v * v;
    __syncthreads();
    for (int off = 32; off > 0; off >>= 1) {
        if (j < off) red[j] += red[j + off];
        __syncthreads();
    }
    k[base + j] = v * rsqrtf(red[0] + 1e-6f);
}

// ---------------------------------------------------------------------------
// Delta-rule scan. One block (64 thr) per (b,h); thread j owns state column
// S[0..63][j] in 64 VGPRs. k/q/g broadcast through LDS each step.
// ---------------------------------------------------------------------------
__global__ __launch_bounds__(64)
void scan_kernel(const float* __restrict__ q, const float* __restrict__ k,
                 const float* __restrict__ v, const float* __restrict__ beta,
                 const float* __restrict__ g, float* __restrict__ o,
                 int S, int H) {
    const int bh = blockIdx.x;
    const int b = bh / H, h = bh % H;
    const int j = threadIdx.x;

    __shared__ float kS[64], qS[64], gS[64];
    __shared__ float bsh;

    float st[64];
#pragma unroll
    for (int i = 0; i < 64; i++) st[i] = 0.f;

    for (int t = 0; t < S; t++) {
        const size_t roff = ((size_t)(b * S + t)) * (H * 64) + h * 64;
        __syncthreads();
        kS[j] = k[roff + j];
        qS[j] = q[roff + j];
        gS[j] = g[roff + j];
        if (j == 0) bsh = beta[(size_t)(b * S + t) * H + h];
        const float vj = v[roff + j];
        __syncthreads();

        float pred = 0.f;
#pragma unroll
        for (int kk = 0; kk < 64; kk++) {
            st[kk] *= gS[kk];                       // per-channel decay
            pred = fmaf(kS[kk], st[kk], pred);      // k^T S
        }
        const float delta = bsh * (vj - pred);
        float oacc = 0.f;
#pragma unroll
        for (int kk = 0; kk < 64; kk++) {
            st[kk] = fmaf(kS[kk], delta, st[kk]);   // rank-1 update
            oacc = fmaf(qS[kk], st[kk], oacc);      // q^T S
        }
        o[roff + j] = oacc;
    }
}

// ---------------------------------------------------------------------------
// Gated RMSNorm over dv=64 + silu(Og) gate -> bf16 for the Wo GEMM.
// One block (64 thr) per (token, head). og already holds silu(h@Wog).
// ---------------------------------------------------------------------------
__global__ __launch_bounds__(64)
void gated_rms_kernel(const float* __restrict__ o, const float* __restrict__ og,
                      const float* __restrict__ rms_w, bf16* __restrict__ ob) {
    const int j = threadIdx.x;
    const size_t base = (size_t)blockIdx.x * 64;
    const float ov = o[base + j];
    __shared__ float red[64];
    red[j] = ov * ov;
    __syncthreads();
    for (int off = 32; off > 0; off >>= 1) {
        if (j < off) red[j] += red[j + off];
        __syncthreads();
    }
    const float inv = rsqrtf(red[0] * (1.0f / 64.0f) + LN_EPS);
    ob[base + j] = (bf16)(ov * inv * rms_w[j] * og[base + j]);
}

// ---------------------------------------------------------------------------
// Host launcher
// ---------------------------------------------------------------------------
extern "C" void kernel_launch(void* const* d_in, const int* in_sizes, int n_in,
                              void* d_out, int out_size, void* d_ws, size_t ws_size,
                              hipStream_t stream) {
    (void)in_sizes; (void)n_in; (void)out_size; (void)ws_size;

    constexpr int Bb = 2, Ss = 2048, Dd = 1024, Hh = 16, Ff = 4096;
    constexpr int Mt = Bb * Ss;  // 4096 tokens

    const float* x     = (const float*)d_in[0];
    const float* ln1_w = (const float*)d_in[1];
    const float* ln1_b = (const float*)d_in[2];
    const float* ln2_w = (const float*)d_in[3];
    const float* ln2_b = (const float*)d_in[4];
    const float* Wq    = (const float*)d_in[5];
    const float* Wk    = (const float*)d_in[6];
    const float* Wv    = (const float*)d_in[7];
    const float* Wb    = (const float*)d_in[8];
    const float* Wg    = (const float*)d_in[9];
    const float* Wog   = (const float*)d_in[10];
    const float* rms_w = (const float*)d_in[11];
    const float* Wo    = (const float*)d_in[12];
    const float* Wu    = (const float*)d_in[13];
    const float* Wv2   = (const float*)d_in[14];
    const float* Wo2   = (const float*)d_in[15];
    float* out = (float*)d_out;

    // ---- workspace carve-out (256B aligned); total ~165 MB, fits L2 ----
    char* ws = (char*)d_ws;
    size_t off = 0;
    auto take = [&](size_t bytes) -> char* {
        char* p = ws + off;
        off += (bytes + 255) & ~(size_t)255;
        return p;
    };
    bf16* WqB  = (bf16*)take((size_t)Dd * Dd * 2);
    bf16* WkB  = (bf16*)take((size_t)Dd * Dd * 2);
    bf16* WvB  = (bf16*)take((size_t)Dd * Dd * 2);
    bf16* WgB  = (bf16*)take((size_t)Dd * Dd * 2);
    bf16* WogB = (bf16*)take((size_t)Dd * Dd * 2);
    bf16* WoB  = (bf16*)take((size_t)Dd * Dd * 2);
    bf16* WbB  = (bf16*)take((size_t)Dd * Hh * 2);
    bf16* WuB  = (bf16*)take((size_t)Dd * Ff * 2);
    bf16* Wv2B = (bf16*)take((size_t)Dd * Ff * 2);
    bf16* Wo2B = (bf16*)take((size_t)Ff * Dd * 2);
    bf16* hb   = (bf16*)take((size_t)Mt * Dd * 2);   // LN1 out; reused as LN2 out
    bf16* ob   = (bf16*)take((size_t)Mt * Dd * 2);   // gated attn out (bf16)
    float* qf  = (float*)take((size_t)Mt * Dd * 4);
    float* kf  = (float*)take((size_t)Mt * Dd * 4);
    float* vf  = (float*)take((size_t)Mt * Dd * 4);
    float* gf  = (float*)take((size_t)Mt * Dd * 4);
    float* ogf = (float*)take((size_t)Mt * Dd * 4);
    float* betaf = (float*)take((size_t)Mt * Hh * 4);
    float* of  = (float*)take((size_t)Mt * Dd * 4);
    float* x1f = (float*)take((size_t)Mt * Dd * 4);
    // MLP intermediates alias dead q..g fp32 buffers (each 32 MB bf16)
    bf16* ub   = (bf16*)qf;   // spans q+k region
    bf16* glub = (bf16*)vf;   // spans v+g region

    // ---- weights fp32 -> bf16 ----
    auto cvt = [&](const float* src, bf16* dst, size_t n) {
        cvt_f32_bf16<<<2048, 256, 0, stream>>>(src, dst, n);
    };
    cvt(Wq,  WqB,  (size_t)Dd * Dd);
    cvt(Wk,  WkB,  (size_t)Dd * Dd);
    cvt(Wv,  WvB,  (size_t)Dd * Dd);
    cvt(Wg,  WgB,  (size_t)Dd * Dd);
    cvt(Wog, WogB, (size_t)Dd * Dd);
    cvt(Wo,  WoB,  (size_t)Dd * Dd);
    cvt(Wb,  WbB,  (size_t)Dd * Hh);
    cvt(Wu,  WuB,  (size_t)Dd * Ff);
    cvt(Wv2, Wv2B, (size_t)Dd * Ff);
    cvt(Wo2, Wo2B, (size_t)Ff * Dd);

    const dim3 blk(256);
    const dim3 gD(Dd / 128, Mt / 64);   // N=1024 GEMMs
    const dim3 gF(Ff / 128, Mt / 64);   // N=4096 GEMMs

    // ---- attention branch ----
    layernorm_bf16<<<Mt, blk, 0, stream>>>(x, ln1_w, ln1_b, hb);
    gemm_bf16_wmma<1><<<gD, blk, 0, stream>>>(hb, WqB,  qf,  nullptr, nullptr, Mt, Dd, Dd); // silu
    gemm_bf16_wmma<1><<<gD, blk, 0, stream>>>(hb, WkB,  kf,  nullptr, nullptr, Mt, Dd, Dd); // silu
    gemm_bf16_wmma<0><<<gD, blk, 0, stream>>>(hb, WvB,  vf,  nullptr, nullptr, Mt, Dd, Dd);
    gemm_bf16_wmma<2><<<gD, blk, 0, stream>>>(hb, WgB,  gf,  nullptr, nullptr, Mt, Dd, Dd); // sigmoid
    gemm_bf16_wmma<1><<<gD, blk, 0, stream>>>(hb, WogB, ogf, nullptr, nullptr, Mt, Dd, Dd); // silu
    beta_kernel<<<(Mt * Hh + 255) / 256, blk, 0, stream>>>(hb, WbB, betaf, Mt * Hh);
    l2norm_kernel<<<Mt * Hh, 64, 0, stream>>>(kf);
    scan_kernel<<<Bb * Hh, 64, 0, stream>>>(qf, kf, vf, betaf, gf, of, Ss, Hh);
    gated_rms_kernel<<<Mt * Hh, 64, 0, stream>>>(of, ogf, rms_w, ob);
    gemm_bf16_wmma<5><<<gD, blk, 0, stream>>>(ob, WoB, x1f, nullptr, x, Mt, Dd, Dd); // +residual

    // ---- MLP branch ----
    layernorm_bf16<<<Mt, blk, 0, stream>>>(x1f, ln2_w, ln2_b, hb);
    gemm_bf16_wmma<3><<<gF, blk, 0, stream>>>(hb, WuB,  nullptr, ub,   nullptr, Mt, Ff, Dd); // u bf16
    gemm_bf16_wmma<4><<<gF, blk, 0, stream>>>(hb, Wv2B, nullptr, glub, ub,      Mt, Ff, Dd); // u*silu
    gemm_bf16_wmma<5><<<gD, blk, 0, stream>>>(glub, Wo2B, out, nullptr, x1f, Mt, Dd, Ff);    // +residual
}
